// ScalarOut_44057774522748
// MI455X (gfx1250) — matrix operand order repaired and checked by reading, not compiled, not run on hardware
//
#include <hip/hip_runtime.h>

typedef __attribute__((ext_vector_type(16))) __bf16 v16bf;
typedef __attribute__((ext_vector_type(8)))  float  v8f;

#define NODE_DIM   128
#define HIDDEN_DIM 64

__device__ __forceinline__ unsigned short f32_to_bf16_bits(float f) {
  union { float f; unsigned u; } v; v.f = f;
  unsigned u = v.u;
  unsigned r = u + 0x7FFFu + ((u >> 16) & 1u);  // round-to-nearest-even
  return (unsigned short)(r >> 16);
}

// Pack hi16(u_even) | hi16(u_odd)<<16 in one v_perm_b32 (truncating f32->bf16).
__device__ __forceinline__ unsigned pack_bf16_trunc(unsigned u_even, unsigned u_odd) {
  return __builtin_amdgcn_perm(u_odd, u_even, 0x07060302u);
}

__global__ void ScalarOut_zero_kernel(float* out, int n) {
  int i = blockIdx.x * blockDim.x + threadIdx.x;
  if (i < n) out[i] = 0.0f;
}

// Main kernel: bf16 WMMA MLP + segment-sum. One 16-node tile per wave per
// grid-stride iteration.
__global__ __launch_bounds__(256) void ScalarOut_mlp_segsum_kernel(
    const float* __restrict__ x, const long long* __restrict__ batch,
    const float* __restrict__ w1, const float* __restrict__ b1,
    const float* __restrict__ w2, const float* __restrict__ b2,
    float* __restrict__ out, int n_tiles)
{
  __shared__ __attribute__((aligned(32))) unsigned short lds_w1[HIDDEN_DIM * NODE_DIM];

  // Cooperative load + f32->bf16 (RNE) convert of w1 into LDS (16 KB), once.
  for (int i = threadIdx.x; i < HIDDEN_DIM * NODE_DIM; i += blockDim.x)
    lds_w1[i] = f32_to_bf16_bits(w1[i]);
  __syncthreads();

  const int lane = threadIdx.x & 31;
  const int wave = threadIdx.x >> 5;
  const int n16  = lane & 15;   // column (h within tile / node row m)
  const int hi   = lane >> 4;   // half-wave

  // Preload B fragments: B[kc][t] = w1^T tile, K in [kc*32, kc*32+32),
  // h in [t*16, t*16+16). B layout used here (contiguous K): lanes 0-15 hold K=0..15 (column
  // n16), lanes 16-31 hold K=16..31.
  v16bf B[4][4];
  #pragma unroll
  for (int kc = 0; kc < 4; ++kc) {
    #pragma unroll
    for (int t = 0; t < 4; ++t) {
      const int h     = t * 16 + n16;
      const int kbase = kc * 32 + hi * 16;   // 16 contiguous K values, 32B aligned
      B[kc][t] = *reinterpret_cast<const v16bf*>(&lds_w1[h * NODE_DIM + kbase]);
    }
  }

  // Per-lane epilogue constants for h = t*16 + n16.
  float b1v[4], w2v[4];
  #pragma unroll
  for (int t = 0; t < 4; ++t) {
    b1v[t] = b1[t * 16 + n16];
    w2v[t] = w2[t * 16 + n16];
  }
  const float bias2 = b2[0];

  const int wave_id = blockIdx.x * (blockDim.x >> 5) + wave;
  const int n_waves = gridDim.x * (blockDim.x >> 5);

  for (int tile = wave_id; tile < n_tiles; tile += n_waves) {
    const int node0 = tile * 16;
    const unsigned* xrow =
        reinterpret_cast<const unsigned*>(x + (long long)(node0 + n16) * NODE_DIM);

    // Prefetch next tile's rows (global_prefetch_b8).
    const int ntile = tile + n_waves;
    if (ntile < n_tiles)
      __builtin_prefetch(x + (long long)(ntile * 16 + n16) * NODE_DIM + hi * 64, 0, 1);

    v8f acc[4];
    #pragma unroll
    for (int t = 0; t < 4; ++t) acc[t] = (v8f){0.f,0.f,0.f,0.f,0.f,0.f,0.f,0.f};

    #pragma unroll
    for (int kc = 0; kc < 4; ++kc) {
      // A fragment (16x32 bf16): lane row m=n16; elems 0..15 -> K=kc*32+hi*16+j
      // (contiguous K order, matching B).
      const int koff = kc * 32 + hi * 16;
      const uint4 f0 = *reinterpret_cast<const uint4*>(xrow + koff);
      const uint4 f1 = *reinterpret_cast<const uint4*>(xrow + koff + 4);
      const uint4 f2 = *reinterpret_cast<const uint4*>(xrow + koff + 8);
      const uint4 f3 = *reinterpret_cast<const uint4*>(xrow + koff + 12);

      union { v16bf v; unsigned u[8]; } a;
      a.u[0] = pack_bf16_trunc(f0.x, f0.y);
      a.u[1] = pack_bf16_trunc(f0.z, f0.w);
      a.u[2] = pack_bf16_trunc(f1.x, f1.y);
      a.u[3] = pack_bf16_trunc(f1.z, f1.w);
      a.u[4] = pack_bf16_trunc(f2.x, f2.y);
      a.u[5] = pack_bf16_trunc(f2.z, f2.w);
      a.u[6] = pack_bf16_trunc(f3.x, f3.y);
      a.u[7] = pack_bf16_trunc(f3.z, f3.w);

      #pragma unroll
      for (int t = 0; t < 4; ++t)
        acc[t] = __builtin_amdgcn_wmma_f32_16x16x32_bf16(
            false, a.v, false, B[kc][t], (short)0, acc[t], false, false);
    }

    // Epilogue: C/D layout -> lane column n16 holds h = t*16+n16 for rows
    // m = v + 8*hi. partial[v] = sum_h silu(acc+b1[h]) * w2[h].
    float partial[8];
    #pragma unroll
    for (int v = 0; v < 8; ++v) {
      float s = 0.f;
      #pragma unroll
      for (int t = 0; t < 4; ++t) {
        const float hv  = acc[t][v] + b1v[t];
        const float sil = hv * __builtin_amdgcn_rcpf(1.0f + __expf(-hv));
        s += sil * w2v[t];
      }
      partial[v] = s;
    }

    // Reduce across the 16 lanes of each half-wave (columns of the h dim).
    #pragma unroll
    for (int v = 0; v < 8; ++v) {
      #pragma unroll
      for (int off = 1; off < 16; off <<= 1)
        partial[v] += __shfl_xor(partial[v], off, 32);
    }

    // One lane per half-wave commits 8 node scalars. batch is sorted: if all
    // 8 rows share a segment, fold to a single atomic.
    if (n16 == 0) {
      const int rbase = node0 + hi * 8;
      const long long s0 = batch[rbase];
      const long long s7 = batch[rbase + 7];
      if (s0 == s7) {
        float s = 0.f;
        #pragma unroll
        for (int v = 0; v < 8; ++v) s += partial[v] + bias2;
        atomicAdd(&out[(int)s0], s);
      } else {
        #pragma unroll
        for (int v = 0; v < 8; ++v)
          atomicAdd(&out[(int)batch[rbase + v]], partial[v] + bias2);
      }
    }
  }
}

// Scalar fallback for a (non-existent for N=1M) tail of <16 nodes.
__global__ void ScalarOut_tail_kernel(
    const float* __restrict__ x, const long long* __restrict__ batch,
    const float* __restrict__ w1, const float* __restrict__ b1,
    const float* __restrict__ w2, const float* __restrict__ b2,
    float* __restrict__ out, int start, int n_nodes)
{
  const int node = start + blockIdx.x;
  if (node >= n_nodes) return;
  __shared__ float red[HIDDEN_DIM];
  const int h = threadIdx.x;
  const float* xr = x + (long long)node * NODE_DIM;
  const float* wr = w1 + h * NODE_DIM;
  float acc = b1[h];
  for (int k = 0; k < NODE_DIM; ++k) acc += xr[k] * wr[k];
  const float sil = acc * __builtin_amdgcn_rcpf(1.0f + __expf(-acc));
  red[h] = sil * w2[h];
  __syncthreads();
  if (h == 0) {
    float s = b2[0];
    for (int i = 0; i < HIDDEN_DIM; ++i) s += red[i];
    atomicAdd(&out[(int)batch[node]], s);
  }
}

extern "C" void kernel_launch(void* const* d_in, const int* in_sizes, int n_in,
                              void* d_out, int out_size, void* d_ws, size_t ws_size,
                              hipStream_t stream) {
  const float*     x     = (const float*)d_in[0];
  const long long* batch = (const long long*)d_in[1];
  const float*     w1    = (const float*)d_in[2];
  const float*     b1    = (const float*)d_in[3];
  const float*     w2    = (const float*)d_in[4];
  const float*     b2    = (const float*)d_in[5];
  float* out = (float*)d_out;

  const int n_nodes = in_sizes[0] / NODE_DIM;
  const int n_tiles = n_nodes / 16;
  const int rem     = n_nodes - n_tiles * 16;

  ScalarOut_zero_kernel<<<(out_size + 255) / 256, 256, 0, stream>>>(out, out_size);

  if (n_tiles > 0) {
    int blocks = 2048;
    if (blocks * 8 > n_tiles) blocks = (n_tiles + 7) / 8;
    if (blocks < 1) blocks = 1;
    ScalarOut_mlp_segsum_kernel<<<blocks, 256, 0, stream>>>(
        x, batch, w1, b1, w2, b2, out, n_tiles);
  }
  if (rem > 0) {
    ScalarOut_tail_kernel<<<rem, HIDDEN_DIM, 0, stream>>>(
        x, batch, w1, b1, w2, b2, out, n_tiles * 16, n_nodes);
  }
}
